// Str2Str_40905268527417
// MI455X (gfx1250) — matrix operand order repaired and checked by reading, not compiled, not run on hardware
//
#include <hip/hip_runtime.h>
#include <hip/hip_bf16.h>

#define LSEQ 768
#define KNBR 128
#define DMSA 256
#define DPAIR 128
#define DSTATE 16
#define L0IN 32
#define DHMP 64
#define DHSC 128
#define SCROWS 64

typedef __attribute__((ext_vector_type(16))) __bf16 v16bf;
typedef __attribute__((ext_vector_type(8))) float v8f;
typedef __attribute__((ext_vector_type(8))) unsigned short u16x8;
typedef __attribute__((ext_vector_type(16))) unsigned short u16x16;

__device__ __forceinline__ unsigned short f2bf(float f) {
  unsigned u = __builtin_bit_cast(unsigned, f);
  return (unsigned short)((u + 0x7FFFu + ((u >> 16) & 1u)) >> 16);
}
__device__ __forceinline__ float bf2f(unsigned short u) {
  return __builtin_bit_cast(float, (unsigned)u << 16);
}

__device__ __forceinline__ v16bf make_frag(u16x8 lo, u16x8 hi) {
  u16x16 c;
#pragma unroll
  for (int j = 0; j < 8; ++j) { c[j] = lo[j]; c[8 + j] = hi[j]; }
  return __builtin_bit_cast(v16bf, c);
}

// A operand 16x32 bf16 (row-major [m][k] source): lane<16 -> row m, K {k0..k0+7, k0+16..k0+23};
// lane>=16 -> same row set, K {k0+8..k0+15, k0+24..k0+31}  (ISA 7.12.2)
__device__ __forceinline__ v16bf load_a_frag(const unsigned short* base, int stride,
                                             int m0, int k0, int lane) {
  int m = m0 + (lane & 15);
  int kb = k0 + ((lane >> 4) << 3);
  const unsigned short* p = base + m * stride + kb;
  u16x8 lo = *(const u16x8*)p;
  u16x8 hi = *(const u16x8*)(p + 16);
  return make_frag(lo, hi);
}

// B operand 32x16 bf16 from transposed weights [n][k]: lane -> column n, K {kb..kb+7, kb+16..kb+23}, kb = k0 + 8*(lane>=16)
__device__ __forceinline__ v16bf load_b_frag(const unsigned short* baseT, int stride,
                                             int n0, int k0, int lane) {
  int n = n0 + (lane & 15);
  int kb = k0 + ((lane >> 4) << 3);
  const unsigned short* p = baseT + n * stride + kb;
  u16x8 lo = *(const u16x8*)p;
  u16x8 hi = *(const u16x8*)(p + 16);
  return make_frag(lo, hi);
}

__device__ __forceinline__ void store_c(float* base, int stride, int m0, int n0, int lane, v8f c) {
  int n = n0 + (lane & 15);
  int mb = m0 + ((lane >> 4) << 3);
#pragma unroll
  for (int r = 0; r < 8; ++r) base[(mb + r) * stride + n] = c[r];
}

// MODE 0: plain f32 store; MODE 1: +bias, relu, bf16 store; MODE 2: f32 += acc + bias
template <int NT, int KS, int MODE>
__device__ __forceinline__ void gemm_block(const unsigned short* A, int as,
                                           const unsigned short* BT, int bs,
                                           const float* bias, void* Cv, int cs,
                                           int wave, int lane) {
  v8f acc[NT] = {};
#pragma unroll
  for (int ks = 0; ks < KS; ++ks) {
    v16bf a = load_a_frag(A, as, wave * 16, ks * 32, lane);
#pragma unroll
    for (int nt = 0; nt < NT; ++nt) {
      v16bf b = load_b_frag(BT, bs, nt * 16, ks * 32, lane);
      acc[nt] = __builtin_amdgcn_wmma_f32_16x16x32_bf16(false, a, false, b,
                                                        (short)0, acc[nt], false, false);
    }
  }
  int mb = wave * 16 + ((lane >> 4) << 3);
#pragma unroll
  for (int nt = 0; nt < NT; ++nt) {
    int n = nt * 16 + (lane & 15);
    if constexpr (MODE == 0) {
      float* C = (float*)Cv;
#pragma unroll
      for (int r = 0; r < 8; ++r) C[(mb + r) * cs + n] = acc[nt][r];
    } else if constexpr (MODE == 1) {
      unsigned short* C = (unsigned short*)Cv;
      float bv = bias[n];
#pragma unroll
      for (int r = 0; r < 8; ++r) {
        float v = acc[nt][r] + bv;
        C[(mb + r) * cs + n] = f2bf(v > 0.f ? v : 0.f);
      }
    } else {
      float* C = (float*)Cv;
      float bv = bias[n];
#pragma unroll
      for (int r = 0; r < 8; ++r) C[(mb + r) * cs + n] += acc[nt][r] + bv;
    }
  }
}

// ---------------- weight prep: f32 [K][N] -> bf16 transposed [Ndst][Kdst], zero-padded ---------
__global__ void k_prep_wT(const float* __restrict__ w, unsigned short* __restrict__ dst,
                          int Ksrc, int Nsrc, int Kdst, int Ndst) {
  int tgt = blockIdx.x * blockDim.x + threadIdx.x;
  if (tgt >= Kdst * Ndst) return;
  int n = tgt / Kdst, k = tgt % Kdst;
  float v = (n < Nsrc && k < Ksrc) ? w[(size_t)k * Nsrc + n] : 0.f;
  dst[(size_t)n * Kdst + k] = f2bf(v);
}

// ---------------- node embedding + sc s0 LN -----------------------------------------------------
__global__ void __launch_bounds__(256) k_node_embed(
    const float* __restrict__ msa, const float* __restrict__ state,
    const float* __restrict__ g_msa, const float* __restrict__ b_msa,
    const float* __restrict__ g_st, const float* __restrict__ b_st,
    const float* __restrict__ g_s0, const float* __restrict__ b_s0,
    const float* __restrict__ w_ex, const float* __restrict__ b_ex,
    const float* __restrict__ g_nd, const float* __restrict__ b_nd,
    unsigned short* __restrict__ node_bf, unsigned short* __restrict__ s0ln_bf) {
  int i = blockIdx.x;
  int t = threadIdx.x;
  __shared__ float red[256];
  __shared__ float xbuf[DMSA];
  __shared__ float sst[DSTATE];
  __shared__ float sst2[DSTATE];
  __shared__ float sout[L0IN];

  float x = msa[(size_t)i * DMSA + t];
  red[t] = x;
  __syncthreads();
  for (int s = 128; s > 0; s >>= 1) { if (t < s) red[t] += red[t + s]; __syncthreads(); }
  float mu = red[0] / (float)DMSA;
  __syncthreads();
  float dv = x - mu;
  red[t] = dv * dv;
  __syncthreads();
  for (int s = 128; s > 0; s >>= 1) { if (t < s) red[t] += red[t + s]; __syncthreads(); }
  float rs = rsqrtf(red[0] / (float)DMSA + 1e-5f);
  __syncthreads();
  float xn = dv * rs;
  xbuf[t] = xn * g_msa[t] + b_msa[t];
  s0ln_bf[(size_t)i * DMSA + t] = f2bf(xn * g_s0[t] + b_s0[t]);
  if (t < DSTATE) sst[t] = state[(size_t)i * DSTATE + t];
  __syncthreads();
  if (t < DSTATE) {
    float m = 0.f;
    for (int c = 0; c < DSTATE; ++c) m += sst[c];
    m /= (float)DSTATE;
    float v = 0.f;
    for (int c = 0; c < DSTATE; ++c) { float d = sst[c] - m; v += d * d; }
    float r2 = rsqrtf(v / (float)DSTATE + 1e-5f);
    sst2[t] = (sst[t] - m) * r2 * g_st[t] + b_st[t];
  }
  __syncthreads();
  // embed_x 272->32: 8 chunks of 34 inputs per output
  {
    int o = t & 31, ch = t >> 5;
    float acc = 0.f;
    for (int f = ch * 34; f < ch * 34 + 34; ++f) {
      float xv = (f < DMSA) ? xbuf[f] : sst2[f - DMSA];
      acc += xv * w_ex[f * L0IN + o];
    }
    red[t] = acc;
  }
  __syncthreads();
  if (t < L0IN) {
    float s = b_ex[t];
    for (int c = 0; c < 8; ++c) s += red[c * 32 + t];
    sout[t] = s;
  }
  __syncthreads();
  if (t < L0IN) {
    float m = 0.f;
    for (int c = 0; c < L0IN; ++c) m += sout[c];
    m /= (float)L0IN;
    float v = 0.f;
    for (int c = 0; c < L0IN; ++c) { float d = sout[c] - m; v += d * d; }
    float r2 = rsqrtf(v / (float)L0IN + 1e-5f);
    node_bf[(size_t)i * L0IN + t] = f2bf((sout[t] - m) * r2 * g_nd[t] + b_nd[t]);
  }
}

// ---------------- exact top-k(128) via rank counting --------------------------------------------
__global__ void __launch_bounds__(256) k_topk(const float* __restrict__ xyz,
                                              int* __restrict__ nbr,
                                              float* __restrict__ relhat,
                                              float* __restrict__ dnbr) {
  int i = blockIdx.x;
  int t = threadIdx.x;
  __shared__ float ca[LSEQ * 3];
  __shared__ float dist[LSEQ];
  for (int e = t; e < LSEQ * 3; e += 256) {
    int j = e / 3, d = e % 3;
    ca[e] = xyz[(size_t)j * 9 + 3 + d];
  }
  __syncthreads();
  float cix = ca[i * 3 + 0], ciy = ca[i * 3 + 1], ciz = ca[i * 3 + 2];
  float dj[3];
  int jj[3];
#pragma unroll
  for (int m = 0; m < 3; ++m) {
    int j = t + m * 256;
    jj[m] = j;
    float dx = ca[j * 3 + 0] - cix, dy = ca[j * 3 + 1] - ciy, dz = ca[j * 3 + 2] - ciz;
    float d = sqrtf(dx * dx + dy * dy + dz * dz + 1e-12f);
    dj[m] = d + (j == i ? 999.9f : 0.f);
    dist[j] = dj[m];
  }
  __syncthreads();
  int rk[3] = {0, 0, 0};
  for (int jp = 0; jp < LSEQ; ++jp) {
    float dp = dist[jp];
#pragma unroll
    for (int m = 0; m < 3; ++m)
      rk[m] += (dp < dj[m]) || (dp == dj[m] && jp < jj[m]);
  }
#pragma unroll
  for (int m = 0; m < 3; ++m) {
    if (rk[m] < KNBR) {
      int j = jj[m];
      int slot = i * KNBR + rk[m];
      nbr[slot] = j;
      float dx = ca[j * 3 + 0] - cix, dy = ca[j * 3 + 1] - ciy, dz = ca[j * 3 + 2] - ciz;
      float d = sqrtf(dx * dx + dy * dy + dz * dz + 1e-12f);
      dnbr[slot] = d;
      float inv = 1.f / (d + 1e-8f);
      relhat[(size_t)slot * 3 + 0] = dx * inv;
      relhat[(size_t)slot * 3 + 1] = dy * inv;
      relhat[(size_t)slot * 3 + 2] = dz * inv;
    }
  }
}

// ---------------- edge embed + message GEMM + frame update (per residue) ------------------------
__global__ void __launch_bounds__(256) k_edge_msg(
    const float* __restrict__ pair, const float* __restrict__ xyz,
    const int* __restrict__ nbr, const float* __restrict__ relhat,
    const float* __restrict__ dnbr, const unsigned short* __restrict__ node_bf,
    const float* __restrict__ g_pr, const float* __restrict__ b_pr,
    const float* __restrict__ b_e1, const float* __restrict__ g_ed1, const float* __restrict__ b_ed1,
    const float* __restrict__ b_e2, const float* __restrict__ g_ed2, const float* __restrict__ b_ed2,
    const unsigned short* __restrict__ wT_e1, const unsigned short* __restrict__ wT_e2,
    const unsigned short* __restrict__ wT_w1, const float* __restrict__ b_w1,
    const float* __restrict__ w_s, const float* __restrict__ b_s,
    const float* __restrict__ w_g, const float* __restrict__ b_g,
    const float* __restrict__ w_v,
    float* __restrict__ out_xyz, float* __restrict__ out_state) {
  int i = blockIdx.x;
  int t = threadIdx.x;
  int lane = t & 31, wave = t >> 5;  // 8 waves <-> 8 M-tiles of 128 rows

  __shared__ int sNbr[KNBR];
  __shared__ float sRel[KNBR * 3];
  __shared__ float sDn[KNBR];
  __shared__ float sMu[KNBR], sRs[KNBR];
  __shared__ __align__(32) unsigned short sChunk[KNBR * 32];   // 8 KB  (pair bf16 K-chunk)
  __shared__ __align__(32) float sTmp[KNBR * 32];              // 16 KB (gemm f32 out / scratch / msg bf16)
  __shared__ __align__(32) unsigned short sA2[KNBR * 96];      // 24 KB (e2 / msg A operand)
  __shared__ float sMean[DHMP];
  __shared__ float sGate[KNBR * 2];
  __shared__ float sOff[6];

  if (t < KNBR) {
    sNbr[t] = nbr[i * KNBR + t];
    sDn[t] = dnbr[i * KNBR + t];
    sRel[t * 3 + 0] = relhat[(size_t)(i * KNBR + t) * 3 + 0];
    sRel[t * 3 + 1] = relhat[(size_t)(i * KNBR + t) * 3 + 1];
    sRel[t * 3 + 2] = relhat[(size_t)(i * KNBR + t) * 3 + 2];
  }
  __syncthreads();

  // pass 1: LN statistics of gathered pair rows (two threads per row)
  int r2t = t & 127, h2t = t >> 7;
  {
    const float* prow = pair + ((size_t)(i * LSEQ + sNbr[r2t])) * DPAIR + h2t * 64;
    float s = 0.f, q = 0.f;
    for (int c = 0; c < 64; ++c) { float v = prow[c]; s += v; q += v * v; }
    sTmp[t] = s;
    sTmp[256 + t] = q;
  }
  __syncthreads();
  if (t < KNBR) {
    float sum = sTmp[t] + sTmp[t + 128];
    float sq = sTmp[256 + t] + sTmp[256 + t + 128];
    float mu = sum / (float)DPAIR;
    float var = sq / (float)DPAIR - mu * mu;
    sMu[t] = mu;
    sRs[t] = rsqrtf(var + 1e-5f);
  }

  // GEMM1: (128 x 128) x (128 x 32), K-chunked bf16 staging of LN'd pair rows
  v8f acc0 = {}, acc1 = {};
  for (int ks = 0; ks < 4; ++ks) {
    __syncthreads();
    {
      const float* prow = pair + ((size_t)(i * LSEQ + sNbr[r2t])) * DPAIR + ks * 32 + h2t * 16;
      float mu = sMu[r2t], rsv = sRs[r2t];
      for (int c = 0; c < 16; ++c) {
        int col = ks * 32 + h2t * 16 + c;
        sChunk[r2t * 32 + h2t * 16 + c] = f2bf((prow[c] - mu) * rsv * g_pr[col] + b_pr[col]);
      }
    }
    __syncthreads();
    v16bf a = load_a_frag(sChunk, 32, wave * 16, 0, lane);
    v16bf b0 = load_b_frag(wT_e1, DPAIR, 0, ks * 32, lane);
    acc0 = __builtin_amdgcn_wmma_f32_16x16x32_bf16(false, a, false, b0, (short)0, acc0, false, false);
    v16bf b1 = load_b_frag(wT_e1, DPAIR, 16, ks * 32, lane);
    acc1 = __builtin_amdgcn_wmma_f32_16x16x32_bf16(false, a, false, b1, (short)0, acc1, false, false);
  }
  __syncthreads();
  store_c(sTmp, 32, wave * 16, 0, lane, acc0);
  store_c(sTmp, 32, wave * 16, 16, lane, acc1);
  __syncthreads();

  // bias + LN (norm_edge1) -> A2 cols [0,32); rbf -> [32,68); seqsep -> 68; zero-pad to 96
  if (t < KNBR) {
    int r = t;
    float buf[32];
    float m = 0.f;
    for (int o = 0; o < 32; ++o) { float v = sTmp[r * 32 + o] + b_e1[o]; buf[o] = v; m += v; }
    m /= 32.f;
    float var = 0.f;
    for (int o = 0; o < 32; ++o) { float d = buf[o] - m; var += d * d; }
    float rr = rsqrtf(var / 32.f + 1e-5f);
    for (int o = 0; o < 32; ++o)
      sA2[r * 96 + o] = f2bf((buf[o] - m) * rr * g_ed1[o] + b_ed1[o]);
    float d = sDn[r];
    const float sg = 20.f / 36.f;
    for (int j = 0; j < 36; ++j) {
      float muj = 20.f * (float)j / 35.f;
      float z = (d - muj) / sg;
      sA2[r * 96 + 32 + j] = f2bf(expf(-z * z));
    }
    int s = sNbr[r] - i;
    sA2[r * 96 + 68] = f2bf((s == 1) ? 1.f : ((s == -1) ? -1.f : 0.f));
    for (int j = 69; j < 96; ++j) sA2[r * 96 + j] = 0;
  }
  __syncthreads();
  gemm_block<2, 3, 0>(sA2, 96, wT_e2, 96, nullptr, sTmp, 32, wave, lane);
  __syncthreads();

  // bias + LN (norm_edge2) -> Amsg cols [64,96); node_i -> [0,32); node_j -> [32,64)
  if (t < KNBR) {
    int r = t;
    float buf[32];
    float m = 0.f;
    for (int o = 0; o < 32; ++o) { float v = sTmp[r * 32 + o] + b_e2[o]; buf[o] = v; m += v; }
    m /= 32.f;
    float var = 0.f;
    for (int o = 0; o < 32; ++o) { float d = buf[o] - m; var += d * d; }
    float rr = rsqrtf(var / 32.f + 1e-5f);
    for (int o = 0; o < 32; ++o)
      sA2[r * 96 + 64 + o] = f2bf((buf[o] - m) * rr * g_ed2[o] + b_ed2[o]);
    for (int o = 0; o < 32; ++o) {
      sA2[r * 96 + o] = node_bf[(size_t)i * L0IN + o];
      sA2[r * 96 + 32 + o] = node_bf[(size_t)sNbr[r] * L0IN + o];
    }
  }
  __syncthreads();
  // msg GEMM (128 x 96)x(96 x 64), fused bias+relu -> bf16 into sTmp region
  unsigned short* sMsgBf = (unsigned short*)sTmp;
  gemm_block<4, 3, 1>(sA2, 96, wT_w1, 96, b_w1, sMsgBf, DHMP, wave, lane);
  __syncthreads();

  // column means over 128 neighbors
  if (t < DHMP) {
    float s = 0.f;
    for (int r = 0; r < KNBR; ++r) s += bf2f(sMsgBf[r * DHMP + t]);
    sMean[t] = s / (float)KNBR;
  }
  __syncthreads();
  if (t < DSTATE) {
    float s = b_s[t];
    for (int c = 0; c < DHMP; ++c) s += sMean[c] * w_s[c * DSTATE + t];
    out_state[(size_t)i * DSTATE + t] = s;
  }
  if (t < KNBR) {
    float g0 = b_g[0], g1 = b_g[1];
    for (int c = 0; c < DHMP; ++c) {
      float mv = bf2f(sMsgBf[t * DHMP + c]);
      g0 += mv * w_g[c * 2 + 0];
      g1 += mv * w_g[c * 2 + 1];
    }
    sGate[t * 2 + 0] = g0;
    sGate[t * 2 + 1] = g1;
  }
  __syncthreads();
  if (t < 6) {
    int c = t / 3, d = t % 3;
    float s = 0.f;
    for (int k = 0; k < KNBR; ++k) s += sGate[k * 2 + c] * sRel[k * 3 + d];
    float cad = xyz[(size_t)i * 9 + 3 + d];
    for (int a = 0; a < 3; ++a) {
      float l1 = xyz[(size_t)i * 9 + a * 3 + d] - cad;
      s += l1 * w_v[c * 3 + a];
    }
    sOff[t] = s;
  }
  __syncthreads();
  if (t == 0) {
    float T0 = sOff[0] * 0.1f, T1 = sOff[1] * 0.1f, T2 = sOff[2] * 0.1f;
    float R0 = sOff[3] * 0.01f, R1 = sOff[4] * 0.01f, R2 = sOff[5] * 0.01f;
    float Qn = sqrtf(1.f + R0 * R0 + R1 * R1 + R2 * R2);
    float qA = 1.f / Qn, qB = R0 / Qn, qC = R1 / Qn, qD = R2 / Qn;
    float r00 = qA * qA + qB * qB - qC * qC - qD * qD;
    float r01 = 2.f * qB * qC - 2.f * qA * qD;
    float r02 = 2.f * qB * qD + 2.f * qA * qC;
    float r10 = 2.f * qB * qC + 2.f * qA * qD;
    float r11 = qA * qA - qB * qB + qC * qC - qD * qD;
    float r12 = 2.f * qC * qD - 2.f * qA * qB;
    float r20 = 2.f * qB * qD - 2.f * qA * qC;
    float r21 = 2.f * qC * qD + 2.f * qA * qB;
    float r22 = qA * qA - qB * qB - qC * qC + qD * qD;
    float cx = xyz[(size_t)i * 9 + 3], cy = xyz[(size_t)i * 9 + 4], cz = xyz[(size_t)i * 9 + 5];
    for (int a = 0; a < 3; ++a) {
      float vx = xyz[(size_t)i * 9 + a * 3 + 0] - cx;
      float vy = xyz[(size_t)i * 9 + a * 3 + 1] - cy;
      float vz = xyz[(size_t)i * 9 + a * 3 + 2] - cz;
      out_xyz[(size_t)i * 9 + a * 3 + 0] = r00 * vx + r01 * vy + r02 * vz + cx + T0;
      out_xyz[(size_t)i * 9 + a * 3 + 1] = r10 * vx + r11 * vy + r12 * vz + cy + T1;
      out_xyz[(size_t)i * 9 + a * 3 + 2] = r20 * vx + r21 * vy + r22 * vz + cz + T2;
    }
  }
}

// ---------------- sidechain head: WMMA MLP chain ------------------------------------------------
__global__ void __launch_bounds__(128) k_sc(
    const unsigned short* __restrict__ s0ln_bf, const float* __restrict__ state_new,
    const unsigned short* __restrict__ wT_s0, const float* __restrict__ b_s0,
    const float* __restrict__ g_nsi, const float* __restrict__ b_nsi,
    const float* __restrict__ w_si, const float* __restrict__ b_si,
    const unsigned short* __restrict__ wT_l1, const float* __restrict__ b_l1,
    const unsigned short* __restrict__ wT_l2, const float* __restrict__ b_l2,
    const unsigned short* __restrict__ wT_l3, const float* __restrict__ b_l3,
    const unsigned short* __restrict__ wT_l4, const float* __restrict__ b_l4,
    const unsigned short* __restrict__ wT_out, const float* __restrict__ b_out,
    float* __restrict__ alpha) {
  int R0 = blockIdx.x * SCROWS;
  int t = threadIdx.x;
  int lane = t & 31, wave = t >> 5;  // 4 waves <-> 4 M-tiles of 64 rows
  __shared__ __align__(32) unsigned short sA[SCROWS * DHSC];  // 16 KB
  __shared__ __align__(32) unsigned short sB[SCROWS * DHSC];  // 16 KB
  __shared__ float sSi[SCROWS * DHSC];                        // 32 KB
  float* sSn = (float*)sB;  // 4 KB overlay, dead before sB's first GEMM use

  // si0 = LN(msa0) @ Ws0   (A operand streamed straight from global bf16)
  gemm_block<8, 8, 0>(s0ln_bf + (size_t)R0 * DMSA, DMSA, wT_s0, DMSA, nullptr, sSi, DHSC, wave, lane);
  if (t < SCROWS) {
    const float* sp = state_new + (size_t)(R0 + t) * DSTATE;
    float x[16];
    float m = 0.f;
    for (int c = 0; c < 16; ++c) { x[c] = sp[c]; m += x[c]; }
    m /= 16.f;
    float v = 0.f;
    for (int c = 0; c < 16; ++c) { float d = x[c] - m; v += d * d; }
    float rr = rsqrtf(v / 16.f + 1e-5f);
    for (int c = 0; c < 16; ++c) sSn[t * 16 + c] = (x[c] - m) * rr * g_nsi[c] + b_nsi[c];
  }
  __syncthreads();
  for (int e = t; e < SCROWS * DHSC; e += 128) {
    int r = e >> 7, c = e & 127;
    float v = sSi[e] + b_s0[c] + b_si[c];
    for (int s = 0; s < 16; ++s) v += sSn[r * 16 + s] * w_si[s * DHSC + c];
    sSi[e] = v;
  }
  __syncthreads();
  // residual block 1
  for (int e = t; e < SCROWS * DHSC; e += 128) { float v = sSi[e]; sA[e] = f2bf(v > 0.f ? v : 0.f); }
  __syncthreads();
  gemm_block<8, 4, 1>(sA, DHSC, wT_l1, DHSC, b_l1, sB, DHSC, wave, lane);
  __syncthreads();
  gemm_block<8, 4, 2>(sB, DHSC, wT_l2, DHSC, b_l2, sSi, DHSC, wave, lane);
  __syncthreads();
  // residual block 2
  for (int e = t; e < SCROWS * DHSC; e += 128) { float v = sSi[e]; sA[e] = f2bf(v > 0.f ? v : 0.f); }
  __syncthreads();
  gemm_block<8, 4, 1>(sA, DHSC, wT_l3, DHSC, b_l3, sB, DHSC, wave, lane);
  __syncthreads();
  gemm_block<8, 4, 2>(sB, DHSC, wT_l4, DHSC, b_l4, sSi, DHSC, wave, lane);
  __syncthreads();
  // output head: 128 -> 20 (N padded to 32), store straight to global
  for (int e = t; e < SCROWS * DHSC; e += 128) { float v = sSi[e]; sA[e] = f2bf(v > 0.f ? v : 0.f); }
  __syncthreads();
  {
    v8f acc[2] = {};
#pragma unroll
    for (int ks = 0; ks < 4; ++ks) {
      v16bf a = load_a_frag(sA, DHSC, wave * 16, ks * 32, lane);
#pragma unroll
      for (int nt = 0; nt < 2; ++nt) {
        v16bf b = load_b_frag(wT_out, DHSC, nt * 16, ks * 32, lane);
        acc[nt] = __builtin_amdgcn_wmma_f32_16x16x32_bf16(false, a, false, b, (short)0, acc[nt], false, false);
      }
    }
    int mb = wave * 16 + ((lane >> 4) << 3);
#pragma unroll
    for (int nt = 0; nt < 2; ++nt) {
      int n = nt * 16 + (lane & 15);
      if (n < 20) {
        float bv = b_out[n];
#pragma unroll
        for (int r = 0; r < 8; ++r)
          alpha[(size_t)(R0 + mb + r) * 20 + n] = acc[nt][r] + bv;
      }
    }
  }
}

// ---------------- launcher ----------------------------------------------------------------------
extern "C" void kernel_launch(void* const* d_in, const int* in_sizes, int n_in,
                              void* d_out, int out_size, void* d_ws, size_t ws_size,
                              hipStream_t stream) {
  (void)in_sizes; (void)n_in; (void)out_size; (void)ws_size;
  const float* msa = (const float*)d_in[0];
  const float* pair = (const float*)d_in[1];
  const float* xyz = (const float*)d_in[2];
  const float* state = (const float*)d_in[3];
  // d_in[4] = idx (arange, unused), d_in[5] = top_k (fixed 128)
  // params flattened in setup_inputs() insertion order:
  //  6 g_msa 7 b_msa 8 g_pair 9 b_pair 10 g_state 11 b_state
  // 12 w_ex 13 b_ex 14 w_e1 15 b_e1 16 w_e2 17 b_e2
  // 18 g_node 19 b_node 20 g_ed1 21 b_ed1 22 g_ed2 23 b_ed2
  // 24 se3.w1.w 25 se3.w1.b 26 se3.ws.w 27 se3.ws.b 28 se3.wg.w 29 se3.wg.b 30 se3.wv
  // 31 sc.ns0.g 32 sc.ns0.b 33 sc.nsi.g 34 sc.nsi.b 35 sc.ls0.w 36 sc.ls0.b
  // 37 sc.lsi.w 38 sc.lsi.b 39..46 sc.l1..l4 (w,b) 47 sc.lout.w 48 sc.lout.b
  auto F = [&](int k) { return (const float*)d_in[k]; };

  char* ws = (char*)d_ws;
  size_t off = 0;
  auto alloc = [&](size_t bytes) -> void* {
    void* p = ws + off;
    off = (off + bytes + 255) & ~(size_t)255;
    return p;
  };
  unsigned short* wT_e1 = (unsigned short*)alloc(32 * 128 * 2);
  unsigned short* wT_e2 = (unsigned short*)alloc(32 * 96 * 2);
  unsigned short* wT_w1 = (unsigned short*)alloc(64 * 96 * 2);
  unsigned short* wT_s0 = (unsigned short*)alloc(128 * 256 * 2);
  unsigned short* wT_l1 = (unsigned short*)alloc(128 * 128 * 2);
  unsigned short* wT_l2 = (unsigned short*)alloc(128 * 128 * 2);
  unsigned short* wT_l3 = (unsigned short*)alloc(128 * 128 * 2);
  unsigned short* wT_l4 = (unsigned short*)alloc(128 * 128 * 2);
  unsigned short* wT_out = (unsigned short*)alloc(32 * 128 * 2);
  unsigned short* node_bf = (unsigned short*)alloc((size_t)LSEQ * L0IN * 2);
  unsigned short* s0ln_bf = (unsigned short*)alloc((size_t)LSEQ * DMSA * 2);
  int* nbr = (int*)alloc((size_t)LSEQ * KNBR * 4);
  float* relhat = (float*)alloc((size_t)LSEQ * KNBR * 3 * 4);
  float* dnbr = (float*)alloc((size_t)LSEQ * KNBR * 4);

  auto prep = [&](const float* w, unsigned short* dst, int K, int N, int Kd, int Nd) {
    int tot = Kd * Nd;
    k_prep_wT<<<(tot + 255) / 256, 256, 0, stream>>>(w, dst, K, N, Kd, Nd);
  };
  prep(F(14), wT_e1, 128, 32, 128, 32);
  prep(F(16), wT_e2, 69, 32, 96, 32);
  prep(F(24), wT_w1, 96, 64, 96, 64);
  prep(F(35), wT_s0, 256, 128, 256, 128);
  prep(F(39), wT_l1, 128, 128, 128, 128);
  prep(F(41), wT_l2, 128, 128, 128, 128);
  prep(F(43), wT_l3, 128, 128, 128, 128);
  prep(F(45), wT_l4, 128, 128, 128, 128);
  prep(F(47), wT_out, 128, 20, 128, 32);

  k_node_embed<<<LSEQ, 256, 0, stream>>>(msa, state, F(6), F(7), F(10), F(11), F(31), F(32),
                                         F(12), F(13), F(18), F(19), node_bf, s0ln_bf);
  k_topk<<<LSEQ, 256, 0, stream>>>(xyz, nbr, relhat, dnbr);

  float* out = (float*)d_out;
  float* out_xyz = out;                      // 768*3*3
  float* out_state = out + LSEQ * 9;         // 768*16
  float* out_alpha = out + LSEQ * 9 + LSEQ * DSTATE;  // 768*10*2

  k_edge_msg<<<LSEQ, 256, 0, stream>>>(pair, xyz, nbr, relhat, dnbr, node_bf,
                                       F(8), F(9), F(15), F(20), F(21), F(17), F(22), F(23),
                                       wT_e1, wT_e2, wT_w1, F(25), F(26), F(27), F(28), F(29),
                                       F(30), out_xyz, out_state);

  k_sc<<<LSEQ / SCROWS, 128, 0, stream>>>(s0ln_bf, out_state, wT_s0, F(36), F(33), F(34),
                                          F(37), F(38), wT_l1, F(40), wT_l2, F(42), wT_l3, F(44),
                                          wT_l4, F(46), wT_out, F(48), out_alpha);
}
